// DLUPack_51488067945158
// MI455X (gfx1250) — matrix operand order repaired and checked by reading, not compiled, not run on hardware
//
#include <hip/hip_runtime.h>
#include <hip/hip_bf16.h>

// ---------------------------------------------------------------------------
// DLUPack / CARAFE-style upsampler for MI455X (gfx1250, wave32, WMMA).
//
// Pipeline (N=4, C=256, H=W=96, CC=64, K=5, SCALE=2):
//   k1: comp = 1x1 conv  -> GEMM Wc(64x256) x X(256x9216)      [WMMA f16, 8 K-steps unrolled]
//   k2: off(8ch)+mask(25ch) 3x3 conv over comp, fused M=48 GEMM [WMMA f16, 9 taps x 2 K-steps]
//   k3: softmax over the 25 mask channels (in place)
//   k4: pixel-shuffle offsets + bilinear grid-sample of mask +
//       CARAFE reformulated as D(256x16) = Xwin(256x64) * B(64x16) [WMMA f16]
//
// Roofline reasoning (23.3 TB/s HBM): k1 is 26 FLOP/byte and k4 writes the
// 151 MB output at ~12.5 FLOP/byte -> both need matrix-core rate (f16 in /
// f32 acc) to stay bandwidth-bound; VALU f32 would be 3-6x off the roofline.
// ---------------------------------------------------------------------------

typedef __attribute__((ext_vector_type(16))) _Float16 v16h;
typedef __attribute__((ext_vector_type(8)))  float    v8f;

union AF { v16h v; _Float16 e[16]; };
union CF { v8f  v; float    e[8];  };

#define NB   4
#define CIN  256
#define HH   96
#define WW   96
#define HW   (HH*WW)       // 9216
#define CC   64
#define H2   192
#define W2   192

__device__ __forceinline__ int klocalA(int j, int hi) {
    // 16-bit A 16x32 layout: lanes 0-15 hold K=0..7,16..23 ; lanes 16-31 hold K=8..15,24..31
    return (j < 8) ? (hi * 8 + j) : (8 + hi * 8 + j);
}

// ---------------------------------------------------------------------------
// Kernel 1: comp[n][cc][p] = sum_c Wc[cc][c] * x[n][c][p] + b[cc]
// One wave per 16x16 output tile; K=256 -> 8 WMMA steps of 32, fully
// unrolled so the compiler can cluster loads and back-to-back the WMMAs.
// ---------------------------------------------------------------------------
__global__ __launch_bounds__(256) void k1_comp_gemm(
    const float* __restrict__ x, const float* __restrict__ wc,
    const float* __restrict__ bc, float* __restrict__ comp)
{
    const int lane = threadIdx.x & 31;
    const int wave = threadIdx.x >> 5;
    const int hi   = lane >> 4;
    const int m    = lane & 15;

    int tile  = blockIdx.x * 8 + wave;        // 9216 wave-tiles total
    int mTile = tile & 3;                      // 4 M tiles (64 ch)
    int rest  = tile >> 2;
    int nT    = rest % 576;                    // 576 spatial tiles of 16
    int nb    = rest / 576;                    // batch

    const int col  = nT * 16 + m;
    const int mRow = mTile * 16 + m;

    CF acc;
    #pragma unroll
    for (int r = 0; r < 8; ++r) acc.e[r] = 0.0f;

    #pragma unroll
    for (int kk = 0; kk < 8; ++kk) {
        AF a, b;
        if (kk + 1 < 8) // stream hint for next K-slab of x (global_prefetch_b8)
            __builtin_prefetch(&x[(nb * CIN + (kk + 1) * 32 + hi * 16) * HW + col], 0, 0);
        #pragma unroll
        for (int j = 0; j < 16; ++j) {
            int kl = klocalA(j, hi);
            a.e[j] = (_Float16)wc[mRow * CIN + kk * 32 + kl];
            int kb = kk * 32 + klocalA(j, hi);
            b.e[j] = (_Float16)x[(nb * CIN + kb) * HW + col];
        }
        acc.v = __builtin_amdgcn_wmma_f32_16x16x32_f16(
            false, a.v, false, b.v, (short)0, acc.v, false, false);
    }

    #pragma unroll
    for (int r = 0; r < 8; ++r) {
        int cc = mTile * 16 + r + hi * 8;
        comp[(nb * CC + cc) * HW + nT * 16 + m] = acc.e[r] + bc[cc];
    }
}

// ---------------------------------------------------------------------------
// Kernel 2: fused off(8) + mask(25) 3x3 conv (pad 1) over comp, as GEMM with
// M padded to 48 (3 M-tiles), K = 9 taps x 64 ch -> 18 WMMA per wave-tile.
// Inner tx/kk unrolled (6 WMMAs static) for load/WMMA overlap.
// ---------------------------------------------------------------------------
__global__ __launch_bounds__(256) void k2_enc_conv(
    const float* __restrict__ comp,
    const float* __restrict__ w_off,  const float* __restrict__ b_off,
    const float* __restrict__ w_mask, const float* __restrict__ b_mask,
    float* __restrict__ offb, float* __restrict__ maskb)
{
    const int lane = threadIdx.x & 31;
    const int wave = threadIdx.x >> 5;
    const int hi   = lane >> 4;
    const int m    = lane & 15;

    int id    = blockIdx.x * 8 + wave;        // 6912 wave-tiles
    int mTile = id % 3;
    int s     = id / 3;                        // 2304 spatial tiles
    int wT    = s % 6;
    int h     = (s / 6) % HH;
    int nb    = s / 576;

    const int w0   = wT * 16;
    const int mRow = mTile * 16 + m;           // 0..47 (33..47 are padding)
    const int colw = w0 + m;

    CF acc;
    #pragma unroll
    for (int r = 0; r < 8; ++r) acc.e[r] = 0.0f;

    for (int ty = 0; ty < 3; ++ty) {
        int hh = h + ty - 1;
        #pragma unroll
        for (int tx = 0; tx < 3; ++tx) {
            #pragma unroll
            for (int kk = 0; kk < 2; ++kk) {
                AF a, b;
                #pragma unroll
                for (int j = 0; j < 16; ++j) {
                    int cA = kk * 32 + klocalA(j, hi);
                    float wv = 0.0f;
                    if (mRow < 8)
                        wv = w_off[((mRow * CC + cA) * 3 + ty) * 3 + tx];
                    else if (mRow < 33)
                        wv = w_mask[(((mRow - 8) * CC + cA) * 3 + ty) * 3 + tx];
                    a.e[j] = (_Float16)wv;

                    int cB = kk * 32 + klocalA(j, hi);
                    int ww = colw + tx - 1;
                    float bv = 0.0f;
                    if (hh >= 0 && hh < HH && ww >= 0 && ww < WW)
                        bv = comp[((nb * CC + cB) * HH + hh) * WW + ww];
                    b.e[j] = (_Float16)bv;
                }
                acc.v = __builtin_amdgcn_wmma_f32_16x16x32_f16(
                    false, a.v, false, b.v, (short)0, acc.v, false, false);
            }
        }
    }

    #pragma unroll
    for (int r = 0; r < 8; ++r) {
        int oc = mTile * 16 + r + hi * 8;
        float v = acc.e[r];
        if (oc < 8)
            offb[((nb * 8 + oc) * HH + h) * WW + colw] = v + b_off[oc];
        else if (oc < 33)
            maskb[((nb * 25 + (oc - 8)) * HH + h) * WW + colw] = v + b_mask[oc - 8];
    }
}

// ---------------------------------------------------------------------------
// Kernel 3: softmax over the 25 mask channels, in place, one thread per pixel.
// ---------------------------------------------------------------------------
__global__ __launch_bounds__(256) void k3_softmax(float* __restrict__ maskb)
{
    int p = blockIdx.x * 256 + threadIdx.x;
    if (p >= NB * HW) return;
    int nb  = p / HW;
    int pix = p % HW;
    float v[25];
    float mx = -1e30f;
    #pragma unroll
    for (int k = 0; k < 25; ++k) {
        v[k] = maskb[(nb * 25 + k) * HW + pix];
        mx = fmaxf(mx, v[k]);
    }
    float ssum = 0.0f;
    #pragma unroll
    for (int k = 0; k < 25; ++k) { v[k] = __expf(v[k] - mx); ssum += v[k]; }
    float inv = 1.0f / ssum;
    #pragma unroll
    for (int k = 0; k < 25; ++k)
        maskb[(nb * 25 + k) * HW + pix] = v[k] * inv;
}

// ---------------------------------------------------------------------------
// Kernel 4: fused pixel-shuffle offsets + bilinear mask sampling + CARAFE.
// Block = one (n, h2, 16-wide w2 tile). B(64x16) built in LDS from sampled
// masks (K = 5 rows x 12 cols, pad->64), then 8 waves x 2 channel-tiles x
// 2 k-steps of WMMA produce the 256x16 output tile.
// ---------------------------------------------------------------------------
__global__ __launch_bounds__(256) void k4_carafe(
    const float* __restrict__ x, const float* __restrict__ offb,
    const float* __restrict__ maskb, float* __restrict__ out)
{
    __shared__ int      s_x0[16], s_x1[16], s_y0[16], s_y1[16];
    __shared__ float    s_wx[16], s_wy[16];
    __shared__ float    s_ms[25][16];
    __shared__ _Float16 s_B[64][16];

    const int t  = threadIdx.x;
    int b    = blockIdx.x;                 // 9216 blocks
    int wT   = b % 12;
    int h2   = (b / 12) % H2;
    int nb   = b / (12 * H2);
    int w2_0 = wT * 16;
    int h    = h2 >> 1;
    int w0lo = w2_0 >> 1;
    int colbase = w0lo - 2;                // PAD = 2

    // Phase A0: per-pixel sampling coordinates (grid = base + shuffled offset)
    if (t < 16) {
        int p  = t;
        int w2 = w2_0 + p;
        int w  = w2 >> 1;
        int sub = (h2 & 1) * 2 + (w2 & 1);
        float offx = offb[((nb * 8 + sub)     * HH + h) * WW + w] * (2.0f / 95.0f);
        float offy = offb[((nb * 8 + 4 + sub) * HH + h) * WW + w] * (2.0f / 95.0f);
        float gx = (-1.0f + (2.0f / 95.0f) * (float)w) + offx;
        float gy = (-1.0f + (2.0f / 95.0f) * (float)h) + offy;
        float sx = fminf(fmaxf((gx + 1.0f) * 0.5f * 95.0f, 0.0f), 95.0f);
        float sy = fminf(fmaxf((gy + 1.0f) * 0.5f * 95.0f, 0.0f), 95.0f);
        float fx = floorf(sx), fy = floorf(sy);
        int x0 = (int)fx, y0 = (int)fy;
        s_x0[p] = x0; s_x1[p] = (x0 + 1 > 95) ? 95 : x0 + 1;
        s_y0[p] = y0; s_y1[p] = (y0 + 1 > 95) ? 95 : y0 + 1;
        s_wx[p] = sx - fx; s_wy[p] = sy - fy;
    }
    __syncthreads();

    // Phase A1: bilinear-sample all 25 softmaxed mask channels for 16 pixels
    for (int idx = t; idx < 400; idx += 256) {
        int k = idx >> 4, p = idx & 15;
        const float* mk = maskb + (nb * 25 + k) * HW;
        float wx = s_wx[p], wy = s_wy[p];
        float v00 = mk[s_y0[p] * WW + s_x0[p]];
        float v01 = mk[s_y0[p] * WW + s_x1[p]];
        float v10 = mk[s_y1[p] * WW + s_x0[p]];
        float v11 = mk[s_y1[p] * WW + s_x1[p]];
        s_ms[k][p] = (v00 * (1.0f - wx) + v01 * wx) * (1.0f - wy)
                   + (v10 * (1.0f - wx) + v11 * wx) * wy;
    }
    __syncthreads();

    // Phase B: build sparse B(64x16) f16: k = dy*12 + (col - colbase)
    for (int e = t; e < 1024; e += 256) {
        int k = e >> 4, p = e & 15;
        float val = 0.0f;
        if (k < 60) {
            int dy  = k / 12;
            int ci  = k - dy * 12;
            int col = colbase + ci;
            int wlo = (w2_0 + p) >> 1;
            int dx  = col - wlo + 2;
            if (dx >= 0 && dx <= 4) val = s_ms[dy * 5 + dx][p];
        }
        s_B[k][p] = (_Float16)val;
    }
    __syncthreads();

    // Phase C: D(16x16 per wave-tile) = Xwin(16x64) * B(64x16) via WMMA
    const int lane = t & 31;
    const int wave = t >> 5;
    const int hi   = lane >> 4;
    const int m    = lane & 15;

    #pragma unroll
    for (int cti = 0; cti < 2; ++cti) {
        int ct = wave + cti * 8;           // 16 channel tiles over 8 waves
        CF acc;
        #pragma unroll
        for (int r = 0; r < 8; ++r) acc.e[r] = 0.0f;

        #pragma unroll
        for (int kk = 0; kk < 2; ++kk) {
            AF a, bf;
            #pragma unroll
            for (int j = 0; j < 16; ++j) {
                int k = kk * 32 + klocalA(j, hi);
                float xv = 0.0f;
                if (k < 60) {
                    int dy  = k / 12;
                    int ci  = k - dy * 12;
                    int row = h + dy - 2;
                    int col = colbase + ci;
                    if (row >= 0 && row < HH && col >= 0 && col < WW)
                        xv = x[((nb * CIN + ct * 16 + m) * HH + row) * WW + col];
                }
                a.e[j]  = (_Float16)xv;
                bf.e[j] = s_B[kk * 32 + klocalA(j, hi)][m];
            }
            acc.v = __builtin_amdgcn_wmma_f32_16x16x32_f16(
                false, a.v, false, bf.v, (short)0, acc.v, false, false);
        }

        #pragma unroll
        for (int r = 0; r < 8; ++r) {
            int c = ct * 16 + r + hi * 8;
            out[((nb * CIN + c) * H2 + h2) * W2 + w2_0 + m] = acc.e[r];
        }
    }
}

// ---------------------------------------------------------------------------
// Host launcher
// ---------------------------------------------------------------------------
extern "C" void kernel_launch(void* const* d_in, const int* in_sizes, int n_in,
                              void* d_out, int out_size, void* d_ws, size_t ws_size,
                              hipStream_t stream) {
    (void)in_sizes; (void)n_in; (void)out_size; (void)ws_size;

    const float* x      = (const float*)d_in[0];
    const float* w_comp = (const float*)d_in[1];
    const float* b_comp = (const float*)d_in[2];
    const float* w_off  = (const float*)d_in[3];
    const float* b_off  = (const float*)d_in[4];
    const float* w_mask = (const float*)d_in[5];
    const float* b_mask = (const float*)d_in[6];
    float* out = (float*)d_out;

    // Workspace layout (floats): comp | off | mask  (~14.3 MB total)
    float* ws    = (float*)d_ws;
    float* comp  = ws;                                   // 4*64*9216
    float* offb  = ws + (size_t)NB * CC * HW;            // 4*8*9216
    float* maskb = offb + (size_t)NB * 8 * HW;           // 4*25*9216

    // k1: 9216 wave-tiles / 8 waves per block
    k1_comp_gemm<<<1152, 256, 0, stream>>>(x, w_comp, b_comp, comp);
    // k2: 6912 wave-tiles / 8
    k2_enc_conv<<<864, 256, 0, stream>>>(comp, w_off, b_off, w_mask, b_mask,
                                         offb, maskb);
    // k3: 36864 pixels
    k3_softmax<<<144, 256, 0, stream>>>(maskb);
    // k4: 4 * 192 * 12 output tiles
    k4_carafe<<<9216, 256, 0, stream>>>(x, offb, maskb, out);
}
